// LSTM_Multivariate_20684562498319
// MI455X (gfx1250) — matrix operand order repaired and checked by reading, not compiled, not run on hardware
//
#include <hip/hip_runtime.h>
#include <hip/hip_bf16.h>
#include <stdint.h>

typedef __attribute__((ext_vector_type(16))) _Float16     v16h;
typedef __attribute__((ext_vector_type(8)))  float        v8f;
typedef __attribute__((ext_vector_type(8)))  unsigned int v8u;

#define FEATS 38
#define HID   64
#define G4    256   // 4*HID gate rows

// ---------------------------------------------------------------------------
// ws layout:
//   [0      , 65536 ) : W1 f16 [256 rows][128 cols]  = [W_ih1 | 0pad | W_hh1]
//   [65536  , 131072) : W2 f16 [256 rows][128 cols]  = [W_ih2 | W_hh2]
//   [131072 , 132096) : bias1 f32[256]  (b_ih1 + b_hh1)
//   [132096 , 133120) : bias2 f32[256]  (b_ih2 + b_hh2)
// ---------------------------------------------------------------------------

__global__ void pack_weights(const float* __restrict__ Wih1, const float* __restrict__ Whh1,
                             const float* __restrict__ bih1, const float* __restrict__ bhh1,
                             const float* __restrict__ Wih2, const float* __restrict__ Whh2,
                             const float* __restrict__ bih2, const float* __restrict__ bhh2,
                             _Float16* __restrict__ W1, _Float16* __restrict__ W2,
                             float* __restrict__ bias1, float* __restrict__ bias2) {
    int j = blockIdx.x;     // gate row 0..255
    int c = threadIdx.x;    // packed K column 0..127
    float w1 = (c < FEATS) ? Wih1[j * FEATS + c]
                           : (c < 64 ? 0.0f : Whh1[j * 64 + (c - 64)]);
    float w2 = (c < 64) ? Wih2[j * 64 + c] : Whh2[j * 64 + (c - 64)];
    W1[j * 128 + c] = (_Float16)w1;
    W2[j * 128 + c] = (_Float16)w2;
    if (c == 0) bias1[j] = bih1[j] + bhh1[j];
    if (c == 1) bias2[j] = bih2[j] + bhh2[j];
}

// CDNA5 V_TANH_F32 transcendental. Trailing v_nop honors the TRANS-op
// result-consumption hazard (compiler can't schedule around inline asm).
__device__ __forceinline__ float tanh_hw(float x) {
    float r;
    asm("v_tanh_f32 %0, %1\n\tv_nop" : "=v"(r) : "v"(x));
    return r;
}
__device__ __forceinline__ float sigm(float x)  { return fmaf(0.5f, tanh_hw(0.5f * x), 0.5f); }

// Build the broadcast A operand (one 32-wide K chunk of the 128-vector,
// replicated into all 16 rows). This code's A k order is k = 16*hi + s (same as B): two
// contiguous 16B runs per lane: u32 pairs [16c+8hi .. +3] and [16c+8hi+4 .. +3].
__device__ __forceinline__ v16h build_a(const _Float16* vec, int chunk, int hi) {
    const uint4* vq = (const uint4*)vec;
    int b = chunk * 4;
    uint4 lo = vq[b + 2 * hi];
    uint4 hx = vq[b + 2 * hi + 1];
    v8u u = {lo.x, lo.y, lo.z, lo.w, hx.x, hx.y, hx.z, hx.w};
    return __builtin_bit_cast(v16h, u);
}

// ---------------------------------------------------------------------------
// Single persistent workgroup, 8 waves (wave32).
//   wq = wid & 3 selects the gate-row quad; half = wid >> 2 selects the K-half
//   (chunks 0..1 vs 2..3 of K=128). Wave quad wq owns gate tiles
//   {wq, wq+4, wq+8, wq+12} -> gates i,f,g,o for k in [16*wq, 16*wq+16), so the
//   gate nonlinearity needs no cross-wave exchange; only the K-halves are
//   reduced through LDS.
// ---------------------------------------------------------------------------
__global__ __launch_bounds__(256, 1)
void lstm2_recurrence(const float* __restrict__ x,
                      const float* __restrict__ h01, const float* __restrict__ c01,
                      const float* __restrict__ h02, const float* __restrict__ c02,
                      const _Float16* __restrict__ W1, const _Float16* __restrict__ W2,
                      const float* __restrict__ bias1, const float* __restrict__ bias2,
                      float* __restrict__ out, int T)
{
    alignas(16) __shared__ _Float16 vec1[2][128]; // [ x(t)(38) | zeros(26) | h1(64) ]
    alignas(16) __shared__ _Float16 vec2[128];    // [ h1(64) | h2(64) ]
    __shared__ float part[G4];                    // K-half partial sums

    const int tid  = threadIdx.x;
    const int wid  = tid >> 5, lane = tid & 31;
    const int wq   = wid & 3, half = wid >> 2;
    const int n    = lane & 15, hi = lane >> 4;
    const int k    = wq * 16 + n;

    // ---- preload B-operand weight tiles into registers (resident all loop) ----
    // B 32x16 f16 in this code's k order (matches A): lane = (n, hi), VGPR v holds K pair (2v + 16*hi)
    v16h w1r[4][2], w2r[4][2];
    {
        const unsigned int* W1u = (const unsigned int*)W1;
        const unsigned int* W2u = (const unsigned int*)W2;
#pragma unroll
        for (int g = 0; g < 4; ++g) {
            int row = (g * 4 + wq) * 16 + n;
#pragma unroll
            for (int c2 = 0; c2 < 2; ++c2) {
                int c = 2 * half + c2;              // global K chunk 0..3
                const unsigned int* p1 = W1u + row * 64 + 16 * c + 8 * hi;
                const unsigned int* p2 = W2u + row * 64 + 16 * c + 8 * hi;
                uint4 a0 = *(const uint4*)p1; uint4 a1 = *(const uint4*)(p1 + 4);
                uint4 b0 = *(const uint4*)p2; uint4 b1 = *(const uint4*)(p2 + 4);
                v8u u1 = {a0.x, a0.y, a0.z, a0.w, a1.x, a1.y, a1.z, a1.w};
                v8u u2 = {b0.x, b0.y, b0.z, b0.w, b1.x, b1.y, b1.z, b1.w};
                w1r[g][c2] = __builtin_bit_cast(v16h, u1);
                w2r[g][c2] = __builtin_bit_cast(v16h, u2);
            }
        }
    }

    // biases + initial cell state (only meaningful in half==0 waves)
    float b1r[4], b2r[4];
#pragma unroll
    for (int g = 0; g < 4; ++g) {
        b1r[g] = bias1[(g * 4 + wq) * 16 + n];
        b2r[g] = bias2[(g * 4 + wq) * 16 + n];
    }
    float c1s = c01[k], c2s = c02[k];

    // ---- initialize LDS state ----
    if (tid < 64) {
        _Float16 h1f = (_Float16)h01[tid];
        vec1[0][64 + tid] = h1f; vec1[1][64 + tid] = h1f;
        vec2[tid]      = h1f;
        vec2[64 + tid] = (_Float16)h02[tid];
    }
    if (tid >= 64 && tid < 90) {            // zero the 26-wide pad, both buffers
        int c = (tid - 64) + FEATS;
        vec1[0][c] = (_Float16)0.0f; vec1[1][c] = (_Float16)0.0f;
    }
    if (tid < 19) {                         // x(0) -> buffer 0
        vec1[0][2 * tid]     = (_Float16)x[2 * tid];
        vec1[0][2 * tid + 1] = (_Float16)x[2 * tid + 1];
    }
    __syncthreads();

    for (int t = 0; t < T; ++t) {
        const int cb = t & 1, nb = cb ^ 1;

        // ================= layer 1 : pre1 = Wcomb1 @ [x_t | 0 | h1] =================
        {
            v16h a0 = build_a(vec1[cb], 2 * half + 0, hi);
            v16h a1 = build_a(vec1[cb], 2 * half + 1, hi);
            v8f acc[4];
#pragma unroll
            for (int g = 0; g < 4; ++g) {
                v8f z = {};
                acc[g] = __builtin_amdgcn_wmma_f32_16x16x32_f16(false, a0, false, w1r[g][0], (short)0, z,      false, false);
                acc[g] = __builtin_amdgcn_wmma_f32_16x16x32_f16(false, a1, false, w1r[g][1], (short)0, acc[g], false, false);
            }
            if (half == 1) {
                if (hi == 0) {
#pragma unroll
                    for (int g = 0; g < 4; ++g) part[(g * 4 + wq) * 16 + n] = acc[g][0];
                }
                // stage x(t+1) into the other buffer while half==0 does gate math
                if (wid == 4 && lane < 19 && (t + 1) < T) {
                    const float* xp = x + (size_t)(t + 1) * FEATS + 2 * lane;
                    vec1[nb][2 * lane]     = (_Float16)xp[0];
                    vec1[nb][2 * lane + 1] = (_Float16)xp[1];
                }
                if (wid == 5 && lane == 0 && (t + 8) < T)
                    __builtin_prefetch(x + (size_t)(t + 8) * FEATS, 0, 0);
            }
            __syncthreads();
            if (half == 0) {
                float pi = acc[0][0] + part[(0 * 4 + wq) * 16 + n] + b1r[0];
                float pf = acc[1][0] + part[(1 * 4 + wq) * 16 + n] + b1r[1];
                float pg = acc[2][0] + part[(2 * 4 + wq) * 16 + n] + b1r[2];
                float po = acc[3][0] + part[(3 * 4 + wq) * 16 + n] + b1r[3];
                c1s = sigm(pf) * c1s + sigm(pi) * tanh_hw(pg);
                float h1 = sigm(po) * tanh_hw(c1s);
                if (hi == 0) {
                    _Float16 hf = (_Float16)h1;
                    vec1[0][64 + k] = hf; vec1[1][64 + k] = hf;
                    vec2[k] = hf;
                }
            }
            __syncthreads();
        }

        // ================= layer 2 : pre2 = [Wih2|Whh2] @ [h1 | h2] =================
        {
            v16h a0 = build_a(vec2, 2 * half + 0, hi);
            v16h a1 = build_a(vec2, 2 * half + 1, hi);
            v8f acc[4];
#pragma unroll
            for (int g = 0; g < 4; ++g) {
                v8f z = {};
                acc[g] = __builtin_amdgcn_wmma_f32_16x16x32_f16(false, a0, false, w2r[g][0], (short)0, z,      false, false);
                acc[g] = __builtin_amdgcn_wmma_f32_16x16x32_f16(false, a1, false, w2r[g][1], (short)0, acc[g], false, false);
            }
            if (half == 1 && hi == 0) {
#pragma unroll
                for (int g = 0; g < 4; ++g) part[(g * 4 + wq) * 16 + n] = acc[g][0];
            }
            __syncthreads();
            if (half == 0) {
                float pi = acc[0][0] + part[(0 * 4 + wq) * 16 + n] + b2r[0];
                float pf = acc[1][0] + part[(1 * 4 + wq) * 16 + n] + b2r[1];
                float pg = acc[2][0] + part[(2 * 4 + wq) * 16 + n] + b2r[2];
                float po = acc[3][0] + part[(3 * 4 + wq) * 16 + n] + b2r[3];
                c2s = sigm(pf) * c2s + sigm(pi) * tanh_hw(pg);
                float h2 = sigm(po) * tanh_hw(c2s);
                if (hi == 0) {
                    vec2[64 + k] = (_Float16)h2;
                    out[(size_t)t * HID + k] = 2.0f * h2;
                }
            }
            __syncthreads();
        }
    }
}

extern "C" void kernel_launch(void* const* d_in, const int* in_sizes, int n_in,
                              void* d_out, int out_size, void* d_ws, size_t ws_size,
                              hipStream_t stream) {
    const float* x    = (const float*)d_in[0];
    const float* h01  = (const float*)d_in[1];
    const float* c01  = (const float*)d_in[2];
    const float* h02  = (const float*)d_in[3];
    const float* c02  = (const float*)d_in[4];
    const float* Wih1 = (const float*)d_in[5];
    const float* Whh1 = (const float*)d_in[6];
    const float* bih1 = (const float*)d_in[7];
    const float* bhh1 = (const float*)d_in[8];
    const float* Wih2 = (const float*)d_in[9];
    const float* Whh2 = (const float*)d_in[10];
    const float* bih2 = (const float*)d_in[11];
    const float* bhh2 = (const float*)d_in[12];
    const int T = in_sizes[0] / FEATS;

    char* ws = (char*)d_ws;
    _Float16* W1 = (_Float16*)ws;
    _Float16* W2 = (_Float16*)(ws + 65536);
    float* bias1 = (float*)(ws + 131072);
    float* bias2 = (float*)(ws + 131072 + 1024);

    pack_weights<<<dim3(G4), dim3(128), 0, stream>>>(Wih1, Whh1, bih1, bhh1,
                                                     Wih2, Whh2, bih2, bhh2,
                                                     W1, W2, bias1, bias2);
    lstm2_recurrence<<<dim3(1), dim3(256), 0, stream>>>(x, h01, c01, h02, c02,
                                                        W1, W2, bias1, bias2,
                                                        (float*)d_out, T);
}
